// Proofreader_26233660244457
// MI455X (gfx1250) — matrix operand order repaired and checked by reading, not compiled, not run on hardware
//
#include <hip/hip_runtime.h>
#include <hip/hip_bf16.h>

typedef __attribute__((ext_vector_type(16))) _Float16 v16h;
typedef __attribute__((ext_vector_type(8)))  _Float16 v8h;
typedef __attribute__((ext_vector_type(8)))  float    v8f;

#define BB 1024
#define SS 5
#define TT 256
#define VV 5
#define HH 128
#define OO 256
#define HP (HH + 8)   // padded LDS row (halves) -> conflict-free b128 loads

__device__ __forceinline__ v8f wmma16(v16h a, v16h b, v8f c) {
    return __builtin_amdgcn_wmma_f32_16x16x32_f16(false, a, false, b, (short)0, c,
                                                  false, false);
}

// B-fragment (32x16 f16): lane holds column n = lane%16, K = c*32 + 16*(lane/16) .. +15
__device__ __forceinline__ v16h bfrag_from_f32(const float* __restrict__ p) {
    v16h r;
#pragma unroll
    for (int i = 0; i < 16; ++i) r[i] = (_Float16)p[i];
    return r;
}

// A-fragment (16x32 f16): lane holds row m = lane%16,
// K = c*32 + 16*(lane/16) + {0..15}
__device__ __forceinline__ v16h afrag_from_h(const _Float16* __restrict__ row, int k0) {
    v8h lo = *(const v8h*)(row + k0 + (k0 & 8));
    v8h hi = *(const v8h*)(row + k0 + (k0 & 8) + 8);
    return __builtin_shufflevector(lo, hi, 0, 1, 2, 3, 4, 5, 6, 7,
                                   8, 9, 10, 11, 12, 13, 14, 15);
}

// Cheap tanh on the serial critical path: prefer the gfx1250 v_tanh_f32 op,
// else exp + fast rcp (no IEEE div expansion).
__device__ __forceinline__ float fast_tanh(float x) {
#if __has_builtin(__builtin_amdgcn_tanhf)
    return __builtin_amdgcn_tanhf(x);
#elif __has_builtin(__builtin_amdgcn_tanh_f32)
    return __builtin_amdgcn_tanh_f32(x);
#else
    float e = __expf(2.0f * x);
    return 1.0f - 2.0f * __builtin_amdgcn_rcpf(e + 1.0f);
#endif
}

// ---------------------------------------------------------------------------
// Kernel 1: the 262,144-step serial sub-RNN chain. One block, 8 waves.
// State [16 x 128] f16 in double-buffered LDS; each wave owns 16 N-columns.
// ---------------------------------------------------------------------------
__global__ __launch_bounds__(256) void subrnn_chain_kernel(
    const float* __restrict__ x,       // [B,S,T,V]
    const float* __restrict__ Wih,     // [H,V]
    const float* __restrict__ Whh,     // [H,H]
    const float* __restrict__ bih,     // [H]
    const float* __restrict__ bhh,     // [H]
    _Float16* __restrict__ sub_out)    // [B,S,H] f16
{
    __shared__ __align__(16) _Float16 hsh[2][16][HP];

    const int tid  = threadIdx.x;
    const int wave = tid >> 5;
    const int lane = tid & 31;
    const int nl   = lane & 15;
    const int hf   = lane >> 4;
    const int n    = wave * 16 + nl;      // output column owned by this lane

    // Preload recurrent weight B-fragments (Whh^T tile), f32 -> f16, once.
    v16h bW[4];
#pragma unroll
    for (int c = 0; c < 4; ++c) {
        float tmp[16];
        const int k0 = c * 32 + 16 * hf;
#pragma unroll
        for (int i = 0; i < 16; ++i) tmp[i] = Whh[n * HH + k0 + i];
        bW[c] = bfrag_from_f32(tmp);
    }
    float wih[VV];
#pragma unroll
    for (int v = 0; v < VV; ++v) wih[v] = Wih[n * VV + v];
    const float bias = bih[n] + bhh[n];

    // zero initial hidden state (buffer 0)
    for (int i = tid; i < 16 * HP; i += 256) (&hsh[0][0][0])[i] = (_Float16)0;
    __syncthreads();

    const int total = BB * TT;
    // prefetch x for step 0 (lane-uniform -> scalar loads)
    float xv[SS][VV];
#pragma unroll
    for (int s = 0; s < SS; ++s)
#pragma unroll
        for (int v = 0; v < VV; ++v) xv[s][v] = x[s * TT * VV + v];

    int buf = 0;
#pragma unroll 1
    for (int step = 0; step < total; ++step) {
        const int b = step >> 8;         // TT == 256
        const int t = step & (TT - 1);

        // issue scalar loads for next step's x (hide SMEM latency behind WMMA)
        float xn[SS][VV];
        {
            const int ns = (step + 1 < total) ? step + 1 : step;
            const int nb = ns >> 8, nt = ns & (TT - 1);
            const float* xp = x + nb * (SS * TT * VV) + nt * VV;
#pragma unroll
            for (int s = 0; s < SS; ++s)
#pragma unroll
                for (int v = 0; v < VV; ++v) xn[s][v] = xp[s * TT * VV + v];
        }

        // Issue all A-fragment LDS loads up front (8x ds_load_b128 back-to-back)
        const _Float16* hrow = &hsh[buf][nl][0];
        v16h aF0 = afrag_from_h(hrow, 0 * 32 + 8 * hf);
        v16h aF1 = afrag_from_h(hrow, 1 * 32 + 8 * hf);
        v16h aF2 = afrag_from_h(hrow, 2 * 32 + 8 * hf);
        v16h aF3 = afrag_from_h(hrow, 3 * 32 + 8 * hf);

        // accumulator init: bias + x_t @ Wih^T  (K=5 term, pure VALU, co-executes)
        v8f acc0, acc1;
#pragma unroll
        for (int r = 0; r < 8; ++r) { acc0[r] = bias; acc1[r] = 0.0f; }
        if (hf == 0) {
#pragma unroll
            for (int r = 0; r < SS; ++r) {   // rows 0..4 are real sub-sequences
                float a = 0.0f;
#pragma unroll
                for (int v = 0; v < VV; ++v) a += xv[r][v] * wih[v];
                acc0[r] += a;
            }
        }

        // K=128 recurrent matmul: two independent WMMA chains of depth 2
        acc0 = wmma16(aF0, bW[0], acc0);
        acc1 = wmma16(aF1, bW[1], acc1);
        acc0 = wmma16(aF2, bW[2], acc0);
        acc1 = wmma16(aF3, bW[3], acc1);

        // nonlinearity + convert
        _Float16 dh[8];
#pragma unroll
        for (int r = 0; r < 8; ++r) dh[r] = (_Float16)fast_tanh(acc0[r] + acc1[r]);

        // write new state into the other buffer; D layout: (M = r + 8*hf, N = n)
        const int nb2 = buf ^ 1;
#pragma unroll
        for (int r = 0; r < 8; ++r) hsh[nb2][r + 8 * hf][n] = dh[r];

        // emit last-timestep hidden of this sample
        if (t == TT - 1 && hf == 0) {
#pragma unroll
            for (int r = 0; r < SS; ++r)
                sub_out[(b * SS + r) * HH + n] = dh[r];
        }

        __syncthreads();
        buf ^= 1;
#pragma unroll
        for (int s = 0; s < SS; ++s)
#pragma unroll
            for (int v = 0; v < VV; ++v) xv[s][v] = xn[s][v];
    }
}

// ---------------------------------------------------------------------------
// Kernel 2: outer RNN (5 steps) + FC, parallel over 64 batch tiles of 16.
// ---------------------------------------------------------------------------
__global__ __launch_bounds__(256) void outer_rnn_fc_kernel(
    const _Float16* __restrict__ sub_out,  // [B,S,H] f16
    const float* __restrict__ Wih,         // [H,H]
    const float* __restrict__ Whh,         // [H,H]
    const float* __restrict__ bih,
    const float* __restrict__ bhh,
    const float* __restrict__ fcw,         // [O,H]
    const float* __restrict__ fcb,         // [O]
    float* __restrict__ out)               // [B,O]
{
    __shared__ __align__(16) _Float16 hsh[16][HP];

    const int m0   = blockIdx.x * 16;
    const int tid  = threadIdx.x;
    const int wave = tid >> 5;
    const int lane = tid & 31;
    const int nl   = lane & 15;
    const int hf   = lane >> 4;
    const int n    = wave * 16 + nl;

    // preload B-fragments of Wih^T and Whh^T for this wave's N-tile
    v16h bWih[4], bWhh[4];
#pragma unroll
    for (int c = 0; c < 4; ++c) {
        const int k0 = c * 32 + 16 * hf;
        bWih[c] = bfrag_from_f32(Wih + n * HH + k0);
        bWhh[c] = bfrag_from_f32(Whh + n * HH + k0);
    }
    const float bias = bih[n] + bhh[n];

    v16h hA[4];
#pragma unroll
    for (int c = 0; c < 4; ++c)
#pragma unroll
        for (int i = 0; i < 16; ++i) hA[c][i] = (_Float16)0;

#pragma unroll
    for (int s = 0; s < SS; ++s) {
        // xs A-fragments straight from global f16 sub_out (row = batch index)
        const _Float16* xrow = sub_out + ((m0 + nl) * SS + s) * HH;
        v16h xA0 = afrag_from_h(xrow, 0 * 32 + 8 * hf);
        v16h xA1 = afrag_from_h(xrow, 1 * 32 + 8 * hf);
        v16h xA2 = afrag_from_h(xrow, 2 * 32 + 8 * hf);
        v16h xA3 = afrag_from_h(xrow, 3 * 32 + 8 * hf);

        // two independent 4-deep chains: x-projection and recurrent term
        v8f accX, accH;
#pragma unroll
        for (int r = 0; r < 8; ++r) { accX[r] = bias; accH[r] = 0.0f; }
        accX = wmma16(xA0, bWih[0], accX);
        accH = wmma16(hA[0], bWhh[0], accH);
        accX = wmma16(xA1, bWih[1], accX);
        accH = wmma16(hA[1], bWhh[1], accH);
        accX = wmma16(xA2, bWih[2], accX);
        accH = wmma16(hA[2], bWhh[2], accH);
        accX = wmma16(xA3, bWih[3], accX);
        accH = wmma16(hA[3], bWhh[3], accH);

        _Float16 dh[8];
#pragma unroll
        for (int r = 0; r < 8; ++r) dh[r] = (_Float16)fast_tanh(accX[r] + accH[r]);

        __syncthreads();   // prior hA loads from hsh are done
#pragma unroll
        for (int r = 0; r < 8; ++r) hsh[r + 8 * hf][n] = dh[r];
        __syncthreads();

        // D->A layout conversion through LDS
#pragma unroll
        for (int c = 0; c < 4; ++c)
            hA[c] = afrag_from_h(&hsh[nl][0], c * 32 + 8 * hf);
    }

    // FC: [16,128] @ [128,256]; each wave does 2 column tiles of 16
#pragma unroll
    for (int half = 0; half < 2; ++half) {
        const int n2 = (wave + 8 * half) * 16 + nl;
        v16h bF[4];
#pragma unroll
        for (int c = 0; c < 4; ++c)
            bF[c] = bfrag_from_f32(fcw + n2 * HH + c * 32 + 16 * hf);

        v8f acc0, acc1;
        const float fb = fcb[n2];
#pragma unroll
        for (int r = 0; r < 8; ++r) { acc0[r] = fb; acc1[r] = 0.0f; }
        acc0 = wmma16(hA[0], bF[0], acc0);
        acc1 = wmma16(hA[1], bF[1], acc1);
        acc0 = wmma16(hA[2], bF[2], acc0);
        acc1 = wmma16(hA[3], bF[3], acc1);

#pragma unroll
        for (int r = 0; r < 8; ++r)
            out[(m0 + r + 8 * hf) * OO + n2] = acc0[r] + acc1[r];
    }
}

extern "C" void kernel_launch(void* const* d_in, const int* in_sizes, int n_in,
                              void* d_out, int out_size, void* d_ws, size_t ws_size,
                              hipStream_t stream) {
    const float* x        = (const float*)d_in[0];
    const float* W_ih_sub = (const float*)d_in[1];
    const float* W_hh_sub = (const float*)d_in[2];
    const float* b_ih_sub = (const float*)d_in[3];
    const float* b_hh_sub = (const float*)d_in[4];
    const float* W_ih     = (const float*)d_in[5];
    const float* W_hh     = (const float*)d_in[6];
    const float* b_ih     = (const float*)d_in[7];
    const float* b_hh     = (const float*)d_in[8];
    const float* fc_w     = (const float*)d_in[9];
    const float* fc_b     = (const float*)d_in[10];

    _Float16* sub_out = (_Float16*)d_ws;   // [B,S,H] f16 = 1.25 MB
    float*    out     = (float*)d_out;     // [B,O] f32

    subrnn_chain_kernel<<<1, 256, 0, stream>>>(x, W_ih_sub, W_hh_sub,
                                               b_ih_sub, b_hh_sub, sub_out);
    outer_rnn_fc_kernel<<<BB / 16, 256, 0, stream>>>(sub_out, W_ih, W_hh,
                                                     b_ih, b_hh, fc_w, fc_b, out);
}
